// LiftSplatShoot_25615184953703
// MI455X (gfx1250) — compile-verified, matrix operand order checked by reading, not run on hardware
//
#include <hip/hip_runtime.h>
#include <hip/hip_bf16.h>

typedef __attribute__((ext_vector_type(16))) _Float16 v16h;
typedef __attribute__((ext_vector_type(8)))  float    v8f;

#define B_  4
#define N_  6
#define D_  41
#define C_  64
#define FH_ 16
#define FW_ 44
#define NX_ 200
#define NY_ 200

__global__ void lss_zero_kernel(float* __restrict__ out, int n) {
    int i = blockIdx.x * blockDim.x + threadIdx.x;
    if (i < n) out[i] = 0.0f;
}

__device__ inline void inv3x3(const float* m, float* o) {
    float a = m[0], b = m[1], c = m[2];
    float d = m[3], e = m[4], f = m[5];
    float g = m[6], h = m[7], i = m[8];
    float A =  (e * i - f * h);
    float Bc = -(d * i - f * g);
    float Cc =  (d * h - e * g);
    float det = a * A + b * Bc + c * Cc;
    float inv = 1.0f / det;
    o[0] = A * inv;   o[1] = (c * h - b * i) * inv;  o[2] = (b * f - c * e) * inv;
    o[3] = Bc * inv;  o[4] = (a * i - c * g) * inv;  o[5] = (c * d - a * f) * inv;
    o[6] = Cc * inv;  o[7] = (b * g - a * h) * inv;  o[8] = (a * e - b * d) * inv;
}

// One wave (32 lanes) per camera pixel.
// softmax(depth logits) -> WMMA outer product with features -> atomic scatter to BEV.
__global__ __launch_bounds__(32) void lss_splat_kernel(
    const float* __restrict__ feat,     // (BN, D+C, FH, FW)
    const float* __restrict__ rots,     // (B, N, 3, 3)
    const float* __restrict__ trans,    // (B, N, 3)
    const float* __restrict__ intrins,  // (B, N, 3, 3)
    const float* __restrict__ prots,    // (B, N, 3, 3)
    const float* __restrict__ ptrans,   // (B, N, 3)
    float* __restrict__ out)            // (B, C, NX, NY)
{
    __shared__ float sW[48];                 // softmaxed depth weights, zero padded
    __shared__ float sF[C_];                 // 64 feature channels for this pixel
    __shared__ alignas(16) int sOff[48];     // per-depth BEV offset vx*NY+vy, -1 = invalid
    __shared__ float sPRinv[9], sCmb[9], sPT[3], sTR[3];

    const int lane = threadIdx.x;
    const int pix  = blockIdx.x;              // bn*FH*FW + h*FW + w
    const int bn   = pix / (FH_ * FW_);
    const int hw   = pix % (FH_ * FW_);
    const int h    = hw / FW_;
    const int w    = hw % FW_;
    const int b    = bn / N_;
    const int n    = bn % N_;

    // ---- per-camera matrices (24 total; one lane per block does 3x3 inverses) ----
    if (lane == 0) {
        float Iinv[9], Pinv[9];
        const float* intr = intrins + (b * N_ + n) * 9;
        const float* ro   = rots    + (b * N_ + n) * 9;
        const float* pr   = prots   + (b * N_ + n) * 9;
        inv3x3(intr, Iinv);
        inv3x3(pr, Pinv);
        #pragma unroll
        for (int k = 0; k < 9; k++) sPRinv[k] = Pinv[k];
        // combine = rots @ inv(intrins)
        #pragma unroll
        for (int r = 0; r < 3; r++)
            #pragma unroll
            for (int c = 0; c < 3; c++)
                sCmb[r * 3 + c] = ro[r * 3 + 0] * Iinv[0 * 3 + c]
                                + ro[r * 3 + 1] * Iinv[1 * 3 + c]
                                + ro[r * 3 + 2] * Iinv[2 * 3 + c];
        #pragma unroll
        for (int k = 0; k < 3; k++) {
            sPT[k] = ptrans[(b * N_ + n) * 3 + k];
            sTR[k] = trans [(b * N_ + n) * 3 + k];
        }
    }

    // ---- softmax over D=41 depth logits (lane holds d=lane and d=lane+32) ----
    const int chanStride = FH_ * FW_;                 // 704
    const float* fbase = feat + (size_t)bn * (D_ + C_) * chanStride + h * FW_ + w;
    float l0 = fbase[lane * chanStride];              // lane < 32 < 41: always valid
    float l1 = (lane + 32 < D_) ? fbase[(lane + 32) * chanStride] : -1e30f;
    float mx = fmaxf(l0, l1);
    #pragma unroll
    for (int o = 16; o > 0; o >>= 1) mx = fmaxf(mx, __shfl_xor(mx, o, 32));
    float e0 = __expf(l0 - mx);
    float e1 = (lane + 32 < D_) ? __expf(l1 - mx) : 0.0f;
    float s  = e0 + e1;
    #pragma unroll
    for (int o = 16; o > 0; o >>= 1) s += __shfl_xor(s, o, 32);
    float rs = 1.0f / s;
    sW[lane] = e0 * rs;
    if (lane < 16) sW[lane + 32] = e1 * rs;           // d in [41,48) -> e1==0

    // ---- feature channels into LDS ----
    sF[lane]      = fbase[(D_ + lane) * chanStride];
    sF[lane + 32] = fbase[(D_ + lane + 32) * chanStride];

    __syncthreads();   // matrices + LDS visible

    // ---- geometry: pixel frustum -> ego -> voxel offset, per depth ----
    const float xs = (float)w * (703.0f / 43.0f);
    const float ys = (float)h * 17.0f;
    auto voxOff = [&](int d) -> int {
        if (d >= D_) return -1;
        float px = xs - sPT[0], py = ys - sPT[1], pz = (4.0f + (float)d) - sPT[2];
        float qx = sPRinv[0] * px + sPRinv[1] * py + sPRinv[2] * pz;
        float qy = sPRinv[3] * px + sPRinv[4] * py + sPRinv[5] * pz;
        float qz = sPRinv[6] * px + sPRinv[7] * py + sPRinv[8] * pz;
        float ux = qx * qz, uy = qy * qz, uz = qz;    // [x*z, y*z, z]
        float gx = sCmb[0] * ux + sCmb[1] * uy + sCmb[2] * uz + sTR[0];
        float gy = sCmb[3] * ux + sCmb[4] * uy + sCmb[5] * uz + sTR[1];
        float gz = sCmb[6] * ux + sCmb[7] * uy + sCmb[8] * uz + sTR[2];
        int vx = (int)floorf((gx + 50.0f) * 2.0f);
        int vy = (int)floorf((gy + 50.0f) * 2.0f);
        int vz = (int)floorf((gz + 10.0f) * (1.0f / 20.0f));
        bool ok = (vx >= 0) & (vx < NX_) & (vy >= 0) & (vy < NY_) & (vz == 0);
        return ok ? (vx * NY_ + vy) : -1;
    };
    sOff[lane] = voxOff(lane);
    if (lane < 16) sOff[lane + 32] = voxOff(lane + 32);

    __syncthreads();

    // ---- lift (outer product) via WMMA, scatter straight from the accumulator ----
    const size_t plane = (size_t)NX_ * NY_;
    float* outB = out + (size_t)b * C_ * plane;
    const int  cLane  = lane & 15;
    const int  hiHalf = lane >> 4;       // D-matrix: M = r + 8*hiHalf, N = lane&15
    const bool lo     = (lane < 16);
    const _Float16 hzero = (_Float16)0.0f;

    // Hoist all fragment source values out of the loops (unconditional LDS
    // reads + v_cndmask selects -> no exec-mask branches in fragment build).
    _Float16 hFeat[4];                   // B fragments' K=0 values, per c0 block
    #pragma unroll
    for (int cb = 0; cb < 4; cb++) {
        float fv = sF[cb * 16 + cLane];
        hFeat[cb] = lo ? (_Float16)fv : hzero;
    }
    _Float16 hWt[3];                     // A fragments' K=0 values, per d0 block
    #pragma unroll
    for (int db = 0; db < 3; db++) {
        float wv = sW[db * 16 + cLane];
        hWt[db] = lo ? (_Float16)wv : hzero;
    }

    const int4* offv = (const int4*)sOff;

    #pragma unroll
    for (int db = 0; db < 3; db++) {
        // This lane's 8 depth offsets for the block: two ds_load_b128.
        const int base4 = db * 4 + hiHalf * 2;       // int4 index into sOff
        int4 o0 = offv[base4 + 0];
        int4 o1 = offv[base4 + 1];
        int offs[8] = {o0.x, o0.y, o0.z, o0.w, o1.x, o1.y, o1.z, o1.w};

        v16h afrag = {};
        afrag[0] = hWt[db];

        #pragma unroll
        for (int cb = 0; cb < 4; cb++) {
            v16h bfrag = {};
            bfrag[0] = hFeat[cb];
            v8f acc = {};
            acc = __builtin_amdgcn_wmma_f32_16x16x32_f16(
                false, afrag, false, bfrag, (short)0, acc, false, false);

            float* outC = outB + (size_t)(cb * 16 + cLane) * plane;
            #pragma unroll
            for (int r = 0; r < 8; r++) {
                if (offs[r] >= 0) atomicAdd(outC + offs[r], acc[r]);
            }
        }
    }
}

extern "C" void kernel_launch(void* const* d_in, const int* in_sizes, int n_in,
                              void* d_out, int out_size, void* d_ws, size_t ws_size,
                              hipStream_t stream) {
    const float* feat    = (const float*)d_in[0];
    const float* rots    = (const float*)d_in[1];
    const float* trans   = (const float*)d_in[2];
    const float* intrins = (const float*)d_in[3];
    const float* prots   = (const float*)d_in[4];
    const float* ptrans  = (const float*)d_in[5];
    float* out = (float*)d_out;

    lss_zero_kernel<<<(out_size + 255) / 256, 256, 0, stream>>>(out, out_size);

    const int nblocks = B_ * N_ * FH_ * FW_;   // 16,896 pixels, one wave each
    lss_splat_kernel<<<nblocks, 32, 0, stream>>>(feat, rots, trans, intrins,
                                                 prots, ptrans, out);
}
